// BertSelfAttention_28355374088419
// MI455X (gfx1250) — matrix operand order repaired and checked by reading, not compiled, not run on hardware
//
#include <hip/hip_runtime.h>
#include <hip/hip_bf16.h>

// ---------------------------------------------------------------------------
// BERT self-attention forward for MI455X (gfx1250, wave32, WMMA bf16)
//   B=4, S=2048, HIDDEN=768, HEADS=12, HEAD_DIM=64
// Stage 1: qkv_gemm  : Q/K/V = X*W + b    (bf16 in workspace, [B,H,S,64])
// Stage 2: attention : streaming-softmax flash attention, all-WMMA,
//                      Q/K tiles staged via GLOBAL_LOAD_ASYNC_TO_LDS_B128
// ---------------------------------------------------------------------------

#define HIDDEN 768
#define SEQ    2048
#define NBATCH 4
#define NHEADS 12
#define HDIM   64

typedef __attribute__((ext_vector_type(16))) __bf16 v16bf;
typedef __attribute__((ext_vector_type(8)))  float  v8f;

#if defined(__gfx1250__) && __has_builtin(__builtin_amdgcn_global_load_async_to_lds_b128)
#define USE_ASYNC_LDS 1
#else
#define USE_ASYNC_LDS 0
#endif

// async-copy operand types: int4 vectors in global (AS1) / LDS (AS3)
typedef __attribute__((ext_vector_type(4))) int i32x4;
typedef __attribute__((address_space(1))) i32x4* gvec_t;
typedef __attribute__((address_space(3))) i32x4* lvec_t;

// round-to-nearest-even fp32 -> bf16 (inputs are finite, skip NaN path)
static __device__ __forceinline__ unsigned short f2bf(float f) {
  unsigned u = __builtin_bit_cast(unsigned, f);
  u += 0x7FFFu + ((u >> 16) & 1u);
  return (unsigned short)(u >> 16);
}

// packed pair conversion: bf16(lo=a, hi=b) as one dword
#if defined(__gfx1250__) && __has_builtin(__builtin_amdgcn_cvt_pk_bf16_f32)
typedef __attribute__((ext_vector_type(2))) __bf16 v2bf;
static __device__ __forceinline__ unsigned pack2bf(float a, float b) {
  v2bf t = __builtin_amdgcn_cvt_pk_bf16_f32(a, b);
  return __builtin_bit_cast(unsigned, t);
}
#else
static __device__ __forceinline__ unsigned pack2bf(float a, float b) {
  return (unsigned)f2bf(a) | ((unsigned)f2bf(b) << 16);
}
#endif

// Fragment-order offsets (elements) inside a 512-bf16 (16x32) tile block.
// A-matrix 16x32 (MxK): lane = m + 16*(k/16), e = k%16
static __device__ __forceinline__ int aoff(int m, int k) {
  return (((m & 15) + (((k >> 4) & 1) << 4)) << 4) + (k & 15);
}
// B-matrix 32x16 (KxN): lane = n + 16*(k/16), e = k%16
static __device__ __forceinline__ int boff(int k, int n) {
  return (((n & 15) + (((k >> 4) & 1) << 4)) << 4) + (k & 15);
}

static __device__ __forceinline__ v8f v8f_zero() {
  v8f z = {0.f, 0.f, 0.f, 0.f, 0.f, 0.f, 0.f, 0.f};
  return z;
}

// ---------------------------------------------------------------------------
// Stage 1: X[8192x768] (fp32) * W[768x768] (fp32) + bias -> bf16 [B,H,S,64]
// Block: 256 threads (8 waves), tile M=128 x N=64 (N-tile == one head), K=32.
// Wave grid 4x2: each wave owns a 32x32 sub-tile (2x2 wmma accumulators).
// ---------------------------------------------------------------------------
__global__ __launch_bounds__(256) void qkv_gemm(
    const float* __restrict__ X, const float* __restrict__ W,
    const float* __restrict__ bias, unsigned short* __restrict__ out) {
  alignas(16) __shared__ unsigned short Xs[8][512];  // 8 row-blocks of 16x32
  alignas(16) __shared__ unsigned short Ws[4][512];  // 4 col-blocks of 32x16

  const int tid   = threadIdx.x;
  const int lane  = tid & 31;
  const int wid   = tid >> 5;
  const int waveM = wid >> 1;            // 0..3  (32 rows each)
  const int waveN = wid & 1;             // 0..1  (32 cols each)
  const int Mbase = blockIdx.x * 128;    // over B*S = 8192
  const int head  = blockIdx.y;          // 0..11 -> Nbase = head*64

  v8f acc[2][2];
  acc[0][0] = v8f_zero(); acc[0][1] = v8f_zero();
  acc[1][0] = v8f_zero(); acc[1][1] = v8f_zero();

  float bval[2];
#pragma unroll
  for (int j = 0; j < 2; ++j)
    bval[j] = bias[head * 64 + (waveN * 2 + j) * 16 + (lane & 15)];

  for (int k0 = 0; k0 < HIDDEN; k0 += 32) {
    __syncthreads();
    // --- X tile 128x32 fp32 -> bf16 fragment order (packed dword stores) ---
    {
      const int row0 = tid >> 3;          // 0..31
      const int kr   = (tid & 7) * 4;     // 0,4,...,28
#pragma unroll
      for (int it = 0; it < 4; ++it) {
        const int r = row0 + it * 32;
        const float4 v =
            *(const float4*)(X + (size_t)(Mbase + r) * HIDDEN + k0 + kr);
        unsigned short* p = &Xs[r >> 4][aoff(r & 15, kr)];
        *(unsigned*)(p + 0) = pack2bf(v.x, v.y);
        *(unsigned*)(p + 2) = pack2bf(v.z, v.w);
      }
    }
    // --- W tile 32x64 fp32 -> bf16 fragment order (B layout) ---
    {
      const int kl = tid >> 4;            // 0..15
      const int nr = (tid & 15) * 4;      // 0..60
#pragma unroll
      for (int it = 0; it < 2; ++it) {
        const int kk = kl + it * 16;
        const float4 v =
            *(const float4*)(W + (size_t)(k0 + kk) * HIDDEN + head * 64 + nr);
        const unsigned short h[4] = {f2bf(v.x), f2bf(v.y), f2bf(v.z), f2bf(v.w)};
#pragma unroll
        for (int q = 0; q < 4; ++q) {
          const int n = nr + q;
          Ws[n >> 4][boff(kk, n & 15)] = h[q];
        }
      }
    }
    __syncthreads();

    v16bf a[2], b[2];
#pragma unroll
    for (int i = 0; i < 2; ++i)
      a[i] = ((const v16bf*)Xs[waveM * 2 + i])[lane];
#pragma unroll
    for (int j = 0; j < 2; ++j)
      b[j] = ((const v16bf*)Ws[waveN * 2 + j])[lane];
#pragma unroll
    for (int i = 0; i < 2; ++i)
#pragma unroll
      for (int j = 0; j < 2; ++j)
        acc[i][j] = __builtin_amdgcn_wmma_f32_16x16x32_bf16(
            false, a[i], false, b[j], (short)0, acc[i][j], false, false);
  }

  // writeout -> bf16 [B,H,S,64] (packed conversion, split b16 stores)
  const int chalf = lane >> 4;
  const int nl    = lane & 15;
#pragma unroll
  for (int i = 0; i < 2; ++i) {
#pragma unroll
    for (int j = 0; j < 2; ++j) {
      const int d = (waveN * 2 + j) * 16 + nl;
#pragma unroll
      for (int r = 0; r < 8; r += 2) {
        const int m0 = Mbase + waveM * 32 + i * 16 + r + chalf * 8;
        const unsigned pw =
            pack2bf(acc[i][j][r] + bval[j], acc[i][j][r + 1] + bval[j]);
        const int b0 = m0 >> 11, s0 = m0 & 2047;
        const int b1 = (m0 + 1) >> 11, s1 = (m0 + 1) & 2047;
        out[(((size_t)b0 * NHEADS + head) * SEQ + s0) * HDIM + d] =
            (unsigned short)pw;
        out[(((size_t)b1 * NHEADS + head) * SEQ + s1) * HDIM + d] =
            (unsigned short)(pw >> 16);
      }
    }
  }
}

// ---------------------------------------------------------------------------
// Stage 2: flash attention. Block = 128 threads (4 waves), 64 query rows per
// block (16 per wave), streams 64 keys per iteration. Grid: (S/64, H, B).
// ---------------------------------------------------------------------------
__global__ __launch_bounds__(128) void attention(
    const unsigned short* __restrict__ Qw, const unsigned short* __restrict__ Kw,
    const unsigned short* __restrict__ Vw, const float* __restrict__ mask,
    float* __restrict__ out) {
  alignas(16) __shared__ unsigned short Qs[4][2][512]; // [rowblk][dchunk] A-frags
  alignas(16) __shared__ unsigned short Ks[2][4][512]; // [dchunk][nblk]  B-frags (scores)
  alignas(16) __shared__ unsigned short Vs[2][4][512]; // [kchunk][nblk]  B-frags (ctx)
  alignas(16) __shared__ unsigned short Ps[4][2][512]; // [wave][kchunk]  A-frags (probs)
  __shared__ float Ms[64];

  const int tid   = threadIdx.x;
  const int lane  = tid & 31;
  const int wid   = tid >> 5;          // 0..3 -> owns query rows wid*16..+15
  const int q0    = blockIdx.x * 64;
  const int h     = blockIdx.y;
  const int bb    = blockIdx.z;
  const size_t hb = ((size_t)bb * NHEADS + h) * SEQ * HDIM;
  const int nl    = lane & 15;
  const int chalf = lane >> 4;

  // --- stage Q tile 64x64 bf16 into A-fragment order ---
#if USE_ASYNC_LDS
  {
    const int row = tid >> 1;                     // 0..63
#pragma unroll
    for (int g2 = 0; g2 < 4; ++g2) {
      const int d  = (tid & 1) * 32 + g2 * 8;     // 8-elem (16B) run start
      const int gl = (d & 31) >> 3;               // 0..3 within 32-chunk
      unsigned short* dst =
          &Qs[row >> 4][d >> 5][((row & 15) + 16 * (gl >> 1)) * 16 + 8 * (gl & 1)];
      const unsigned short* src = Qw + hb + (size_t)(q0 + row) * HDIM + d;
      __builtin_amdgcn_global_load_async_to_lds_b128((gvec_t)src, (lvec_t)dst,
                                                     0, 0);
    }
  }
  asm volatile("s_wait_asynccnt 0x0" ::: "memory");
#else
  {
    const int row = tid >> 1;                     // 0..63
#pragma unroll
    for (int s = 0; s < 8; ++s) {
      const int d = (tid & 1) * 32 + s * 4;
      const ushort4 v =
          *(const ushort4*)(Qw + hb + (size_t)(q0 + row) * HDIM + d);
      unsigned short* p = &Qs[row >> 4][d >> 5][aoff(row & 15, d & 31)];
      p[0] = v.x; p[1] = v.y; p[2] = v.z; p[3] = v.w;
    }
  }
#endif
  __syncthreads();
  v16bf aq[2];
  aq[0] = ((const v16bf*)Qs[wid][0])[lane];
  aq[1] = ((const v16bf*)Qs[wid][1])[lane];

  v8f o[4];
#pragma unroll
  for (int nb = 0; nb < 4; ++nb) o[nb] = v8f_zero();
  float rowM[8], rowL[8];
#pragma unroll
  for (int r = 0; r < 8; ++r) { rowM[r] = -1e30f; rowL[r] = 0.f; }

  for (int kb = 0; kb < SEQ; kb += 64) {
    __syncthreads();
    // --- stage K (async fragment-order copy) ---
    {
      const int key = tid >> 1;
#if USE_ASYNC_LDS
#pragma unroll
      for (int g2 = 0; g2 < 4; ++g2) {
        const int d  = (tid & 1) * 32 + g2 * 8;
        const int gl = (d & 31) >> 3;
        unsigned short* dst =
            &Ks[d >> 5][key >> 4]
               [((key & 15) + 16 * (gl >> 1)) * 16 + 8 * (gl & 1)];
        const unsigned short* src = Kw + hb + (size_t)(kb + key) * HDIM + d;
        __builtin_amdgcn_global_load_async_to_lds_b128((gvec_t)src, (lvec_t)dst,
                                                       0, 0);
      }
#else
#pragma unroll
      for (int s = 0; s < 8; ++s) {
        const int d = (tid & 1) * 32 + s * 4;
        const ushort4 v =
            *(const ushort4*)(Kw + hb + (size_t)(kb + key) * HDIM + d);
        unsigned short* p = &Ks[d >> 5][key >> 4][boff(d & 31, key & 15)];
        p[0] = v.x; p[1] = v.y; p[2] = v.z; p[3] = v.w;
      }
#endif
      // --- stage V (reduction dim is LDS-contiguous dim: manual scatter) ---
#pragma unroll
      for (int s = 0; s < 8; ++s) {
        const int d = (tid & 1) * 32 + s * 4;
        const ushort4 v =
            *(const ushort4*)(Vw + hb + (size_t)(kb + key) * HDIM + d);
        const int kc = key >> 5, kl = key & 31;
        Vs[kc][(d + 0) >> 4][boff(kl, (d + 0) & 15)] = v.x;
        Vs[kc][(d + 1) >> 4][boff(kl, (d + 1) & 15)] = v.y;
        Vs[kc][(d + 2) >> 4][boff(kl, (d + 2) & 15)] = v.z;
        Vs[kc][(d + 3) >> 4][boff(kl, (d + 3) & 15)] = v.w;
      }
      if (tid < 64) Ms[tid] = mask[(size_t)bb * SEQ + kb + tid];
      if (kb + 64 < SEQ) {  // prefetch next tiles -> global_prefetch_b8
        __builtin_prefetch(Kw + hb + (size_t)(kb + 64 + (tid >> 1)) * HDIM, 0, 0);
        __builtin_prefetch(Vw + hb + (size_t)(kb + 64 + (tid >> 1)) * HDIM, 0, 0);
      }
    }
#if USE_ASYNC_LDS
    asm volatile("s_wait_asynccnt 0x0" ::: "memory");
#endif
    __syncthreads();

    // --- scores: 16x64 = Q(16x64) x K^T, 2 k-steps per 16x16 tile ---
    v8f c[4];
#pragma unroll
    for (int nb = 0; nb < 4; ++nb) {
      const v16bf b0 = ((const v16bf*)Ks[0][nb])[lane];
      const v16bf b1 = ((const v16bf*)Ks[1][nb])[lane];
      v8f t = v8f_zero();
      t = __builtin_amdgcn_wmma_f32_16x16x32_bf16(false, aq[0], false, b0,
                                                  (short)0, t, false, false);
      t = __builtin_amdgcn_wmma_f32_16x16x32_bf16(false, aq[1], false, b1,
                                                  (short)0, t, false, false);
      c[nb] = t;
    }
    // scale 1/sqrt(64) and add attention mask
#pragma unroll
    for (int nb = 0; nb < 4; ++nb) {
      const float mv = Ms[nb * 16 + nl];
#pragma unroll
      for (int r = 0; r < 8; ++r) c[nb][r] = c[nb][r] * 0.125f + mv;
    }
    // --- online softmax; rows live in lane-halves of the C layout ---
#pragma unroll
    for (int r = 0; r < 8; ++r) {
      float mx = fmaxf(fmaxf(c[0][r], c[1][r]), fmaxf(c[2][r], c[3][r]));
#pragma unroll
      for (int off = 8; off > 0; off >>= 1)
        mx = fmaxf(mx, __shfl_xor(mx, off, 16));
      const float newM  = fmaxf(rowM[r], mx);
      const float scale = __expf(rowM[r] - newM);
      float psum = 0.f;
#pragma unroll
      for (int nb = 0; nb < 4; ++nb) {
        const float p = __expf(c[nb][r] - newM);
        c[nb][r] = p;
        psum += p;
      }
#pragma unroll
      for (int off = 8; off > 0; off >>= 1)
        psum += __shfl_xor(psum, off, 16);
      rowL[r] = rowL[r] * scale + psum;
      rowM[r] = newM;
#pragma unroll
      for (int nb = 0; nb < 4; ++nb) o[nb][r] *= scale;
    }
    // --- probs -> per-wave LDS, A-fragment order (bf16) ---
#pragma unroll
    for (int nb = 0; nb < 4; ++nb) {
      const int keyloc = nb * 16 + nl;    // 0..63
      unsigned short* pd = Ps[wid][keyloc >> 5];
      const int kk = keyloc & 31;
#pragma unroll
      for (int r = 0; r < 8; ++r)
        pd[aoff(r + chalf * 8, kk)] = f2bf(c[nb][r]);
    }
    asm volatile("s_wait_dscnt 0x0" ::: "memory");
    // --- ctx accumulate: O(16x64) += P(16x64) x V(64x64) ---
    const v16bf pa0 = ((const v16bf*)Ps[wid][0])[lane];
    const v16bf pa1 = ((const v16bf*)Ps[wid][1])[lane];
#pragma unroll
    for (int nb = 0; nb < 4; ++nb) {
      const v16bf v0 = ((const v16bf*)Vs[0][nb])[lane];
      const v16bf v1 = ((const v16bf*)Vs[1][nb])[lane];
      o[nb] = __builtin_amdgcn_wmma_f32_16x16x32_bf16(false, pa0, false, v0,
                                                      (short)0, o[nb], false, false);
      o[nb] = __builtin_amdgcn_wmma_f32_16x16x32_bf16(false, pa1, false, v1,
                                                      (short)0, o[nb], false, false);
    }
  }

  // --- normalize and write out fp32 [B,S,768] ---
#pragma unroll
  for (int r = 0; r < 8; ++r) rowL[r] = 1.0f / rowL[r];
#pragma unroll
  for (int nb = 0; nb < 4; ++nb) {
#pragma unroll
    for (int r = 0; r < 8; ++r) {
      const int srow = q0 + wid * 16 + r + chalf * 8;
      out[((size_t)bb * SEQ + srow) * HIDDEN + h * 64 + nb * 16 + nl] =
          o[nb][r] * rowL[r];
    }
  }
}

// ---------------------------------------------------------------------------
extern "C" void kernel_launch(void* const* d_in, const int* in_sizes, int n_in,
                              void* d_out, int out_size, void* d_ws,
                              size_t ws_size, hipStream_t stream) {
  const float* X    = (const float*)d_in[0];
  const float* mask = (const float*)d_in[1];
  const float* Wq   = (const float*)d_in[2];
  const float* bq   = (const float*)d_in[3];
  const float* Wk   = (const float*)d_in[4];
  const float* bk   = (const float*)d_in[5];
  const float* Wv   = (const float*)d_in[6];
  const float* bv   = (const float*)d_in[7];
  float* out        = (float*)d_out;

  const size_t elems = (size_t)NBATCH * SEQ * HIDDEN;  // 6,291,456 bf16 each
  unsigned short* Qw = (unsigned short*)d_ws;
  unsigned short* Kw = Qw + elems;
  unsigned short* Vw = Kw + elems;

  const dim3 g1((NBATCH * SEQ) / 128, NHEADS);  // 64 x 12
  qkv_gemm<<<g1, 256, 0, stream>>>(X, Wq, bq, Qw);
  qkv_gemm<<<g1, 256, 0, stream>>>(X, Wk, bk, Kw);
  qkv_gemm<<<g1, 256, 0, stream>>>(X, Wv, bv, Vw);

  const dim3 g2(SEQ / 64, NHEADS, NBATCH);      // 32 x 12 x 4
  attention<<<g2, 128, 0, stream>>>(Qw, Kw, Vw, mask, out);
}
